// TransformerEncoderCell_56298431316009
// MI455X (gfx1250) — matrix operand order repaired and checked by reading, not compiled, not run on hardware
//
#include <hip/hip_runtime.h>
#include <hip/hip_bf16.h>

typedef __bf16 bf16;
typedef __attribute__((ext_vector_type(16))) __bf16 v16bf;
typedef __attribute__((ext_vector_type(8)))  float  v8f;

// Problem constants
static const int Bn  = 4;
static const int Ln  = 2048;
static const int Dn  = 1024;
static const int HKn = 1024;   // NUM_HEAD * NUM_K
static const int HVn = 1024;   // NUM_HEAD * NUM_V
static const int Hid = 4096;

#if defined(__HIP_DEVICE_COMPILE__) && \
    __has_builtin(__builtin_amdgcn_global_load_async_to_lds_b128) && \
    __has_builtin(__builtin_amdgcn_s_wait_asynccnt)
#define HAVE_ASYNC_LDS 1
#else
#define HAVE_ASYNC_LDS 0
#endif

#if HAVE_ASYNC_LDS
typedef int v4i_t __attribute__((ext_vector_type(4)));
typedef __attribute__((address_space(1))) v4i_t* g_v4i_p;   // global
typedef __attribute__((address_space(3))) v4i_t* l_v4i_p;   // LDS
static __device__ __forceinline__ void async_cp16(const bf16* g, bf16* l) {
  __builtin_amdgcn_global_load_async_to_lds_b128((g_v4i_p)g, (l_v4i_p)l,
                                                 /*offset=*/0, /*cpol=*/0);
}
#endif

// ---------------------------------------------------------------------------
// f32 -> bf16 conversion (4-wide)
// ---------------------------------------------------------------------------
__global__ void cvt_f32_bf16(const float* __restrict__ in, bf16* __restrict__ out,
                             long long n4) {
  long long i = (long long)blockIdx.x * blockDim.x + threadIdx.x;
  long long stride = (long long)gridDim.x * blockDim.x;
  for (; i < n4; i += stride) {
    float4 f = ((const float4*)in)[i];
    union { bf16 h[4]; uint2 u; } o;
    o.h[0] = (bf16)f.x; o.h[1] = (bf16)f.y; o.h[2] = (bf16)f.z; o.h[3] = (bf16)f.w;
    ((uint2*)out)[i] = o.u;
  }
}

// ---------------------------------------------------------------------------
// Transpose + convert: in (R x C, f32) -> out (C x R, bf16)
// block (32,8), grid (C/32, R/32); R,C multiples of 32.
// ---------------------------------------------------------------------------
__global__ __launch_bounds__(256)
void transpose_cvt_f32_bf16(const float* __restrict__ in, bf16* __restrict__ out,
                            int R, int C) {
  __shared__ float t[32][33];
  const int tx = threadIdx.x, ty = threadIdx.y;
  const int c0 = blockIdx.x * 32, r0 = blockIdx.y * 32;
#pragma unroll
  for (int i = 0; i < 4; ++i)
    t[ty + i * 8][tx] = in[(long long)(r0 + ty + i * 8) * C + c0 + tx];
  __syncthreads();
#pragma unroll
  for (int i = 0; i < 4; ++i)
    out[(long long)(c0 + ty + i * 8) * R + r0 + tx] = (bf16)t[tx][ty + i * 8];
}

// ---------------------------------------------------------------------------
// WMMA bf16 GEMM, B always given N-major:  C[M,N] = A[M,K] x B^T(N,K)
//   A: M x K  (leading dim lda)
//   B: N x K  (leading dim ldb)   <- "transposed" storage
//   C: M x N  (leading dim ldc), or C^T: N x M (leading dim ldc) if OUT_TRANS
// Block tile 128x128x64, 256 threads (8 waves, 2x4), wave tile 64x32.
// Double-buffered LDS, async global->LDS DMA when available.
// ---------------------------------------------------------------------------
template<bool OUT_BF16, bool OUT_TRANS, bool HAS_BIAS, bool RELU, bool SCALE>
__global__ __launch_bounds__(256)
void gemm_bt_wmma(const bf16* __restrict__ A, const bf16* __restrict__ Bm,
                  void* __restrict__ Cp, const float* __restrict__ bias,
                  float alpha, int M, int N, int K,
                  int lda, int ldb, int ldc,
                  long long sA, long long sB, long long sC) {
  constexpr int BM = 128, BN = 128, BK = 64, PITCH = 72; // PITCH bf16 (144B rows)
  __shared__ alignas(16) bf16 As[2][BM * PITCH];
  __shared__ alignas(16) bf16 Bs[2][BN * PITCH];

  const int tid  = threadIdx.x;
  const int lane = tid & 31;
  const int wave = tid >> 5;
  const int wm   = (wave >> 2) * 64;
  const int wn   = (wave & 3) * 32;
  const int bm0  = blockIdx.y * BM;
  const int bn0  = blockIdx.x * BN;
  const int l15  = lane & 15;
  const int lhi  = (lane < 16) ? 0 : 1;

  A  += (long long)blockIdx.z * sA;
  Bm += (long long)blockIdx.z * sB;

  v8f acc[4][2];
#pragma unroll
  for (int i = 0; i < 4; ++i)
#pragma unroll
    for (int j = 0; j < 2; ++j)
#pragma unroll
      for (int e = 0; e < 8; ++e) acc[i][j][e] = 0.0f;

#if HAVE_ASYNC_LDS
  auto issue = [&](int k0, int buf) {
#pragma unroll
    for (int it = 0; it < 4; ++it) {
      int c = tid + it * 256;
      int row = c >> 3, cg = c & 7;
      async_cp16(A  + (long long)(bm0 + row) * lda + k0 + cg * 8,
                 &As[buf][row * PITCH + cg * 8]);
    }
#pragma unroll
    for (int it = 0; it < 4; ++it) {
      int c = tid + it * 256;
      int row = c >> 3, cg = c & 7;
      async_cp16(Bm + (long long)(bn0 + row) * ldb + k0 + cg * 8,
                 &Bs[buf][row * PITCH + cg * 8]);
    }
  };
  issue(0, 0);
#else
  auto stage_sync = [&](int k0, int buf) {
    uint4 va[4], vb[4];
#pragma unroll
    for (int it = 0; it < 4; ++it) {
      int c = tid + it * 256;
      int row = c >> 3, cg = c & 7;
      va[it] = *(const uint4*)(A  + (long long)(bm0 + row) * lda + k0 + cg * 8);
      vb[it] = *(const uint4*)(Bm + (long long)(bn0 + row) * ldb + k0 + cg * 8);
    }
#pragma unroll
    for (int it = 0; it < 4; ++it) {
      int c = tid + it * 256;
      int row = c >> 3, cg = c & 7;
      *(uint4*)(&As[buf][row * PITCH + cg * 8]) = va[it];
      *(uint4*)(&Bs[buf][row * PITCH + cg * 8]) = vb[it];
    }
  };
#endif

  const int nsteps = K / BK;
  for (int step = 0; step < nsteps; ++step) {
    const int buf = step & 1;
#if HAVE_ASYNC_LDS
    __builtin_amdgcn_s_wait_asynccnt(0);  // own stage `step` DMA done
    __syncthreads();                      // all waves' data visible; prev compute done
    if (step + 1 < nsteps) issue((step + 1) * BK, buf ^ 1);
#else
    stage_sync(step * BK, buf);
    __syncthreads();
#endif

    // ---- compute two K=32 sub-steps from this stage ----
    union F { v16bf v; uint4 q[2]; };
#pragma unroll
    for (int s = 0; s < 2; ++s) {
      F a[4], b[2];
#pragma unroll
      for (int t = 0; t < 4; ++t) {
        const bf16* p = &As[buf][(wm + t * 16 + l15) * PITCH + s * 32 + lhi * 8];
        a[t].q[0] = *(const uint4*)(p);
        a[t].q[1] = *(const uint4*)(p + 16);
      }
#pragma unroll
      for (int t = 0; t < 2; ++t) {
        const bf16* p = &Bs[buf][(wn + t * 16 + l15) * PITCH + s * 32 + lhi * 8];
        b[t].q[0] = *(const uint4*)(p);
        b[t].q[1] = *(const uint4*)(p + 16);
      }
#pragma unroll
      for (int i = 0; i < 4; ++i)
#pragma unroll
        for (int j = 0; j < 2; ++j)
          acc[i][j] = __builtin_amdgcn_wmma_f32_16x16x32_bf16(
              false, a[i].v, false, b[j].v, (short)0, acc[i][j], false, false);
    }
#if !HAVE_ASYNC_LDS
    __syncthreads();
#endif
  }

  // ---- epilogue ----
  const long long cbat = (long long)blockIdx.z * sC;
#pragma unroll
  for (int i = 0; i < 4; ++i) {
#pragma unroll
    for (int j = 0; j < 2; ++j) {
      int n = bn0 + wn + j * 16 + l15;
      float bv = HAS_BIAS ? bias[n] : 0.0f;
#pragma unroll
      for (int e = 0; e < 8; ++e) {
        int m = bm0 + wm + i * 16 + lhi * 8 + e;
        float v = acc[i][j][e];
        if (SCALE) v *= alpha;
        v += bv;
        if (RELU) v = fmaxf(v, 0.0f);
        long long idx = OUT_TRANS ? (cbat + (long long)n * ldc + m)
                                  : (cbat + (long long)m * ldc + n);
        if (OUT_BF16) ((bf16*)Cp)[idx] = (bf16)v;
        else          ((float*)Cp)[idx] = v;
      }
    }
  }
}

// ---------------------------------------------------------------------------
// Masked softmax over rows of length 2048; writes bf16 attention weights.
// grid = (L, B), block = 256, 8 elements / thread.
// ---------------------------------------------------------------------------
__global__ __launch_bounds__(256)
void softmax_mask_kernel(const float* __restrict__ scores,
                         const unsigned char* __restrict__ mask,
                         bf16* __restrict__ attn) {
  const long long row = (long long)blockIdx.y * gridDim.x + blockIdx.x;
  const float* s = scores + row * Ln;
  const unsigned char* mk = mask + row * Ln;
  bf16* o = attn + row * Ln;
  const int tid = threadIdx.x;

  float vals[8];
  float mx = -__builtin_inff();
#pragma unroll
  for (int i = 0; i < 8; ++i) {
    int idx = tid + i * 256;
    float v = s[idx];
    if (mk[idx]) v = -__builtin_inff();
    vals[i] = v;
    mx = fmaxf(mx, v);
  }
  __shared__ float red[256];
  red[tid] = mx; __syncthreads();
  for (int off = 128; off > 0; off >>= 1) {
    if (tid < off) red[tid] = fmaxf(red[tid], red[tid + off]);
    __syncthreads();
  }
  mx = red[0]; __syncthreads();

  float sum = 0.0f;
#pragma unroll
  for (int i = 0; i < 8; ++i) {
    float e = __expf(vals[i] - mx);
    vals[i] = e;
    sum += e;
  }
  red[tid] = sum; __syncthreads();
  for (int off = 128; off > 0; off >>= 1) {
    if (tid < off) red[tid] += red[tid + off];
    __syncthreads();
  }
  float inv = 1.0f / red[0];
#pragma unroll
  for (int i = 0; i < 8; ++i) o[tid + i * 256] = (bf16)(vals[i] * inv);
}

// ---------------------------------------------------------------------------
// y = LayerNorm(a + r) * gamma + beta ; optionally also emit bf16 copy.
// ---------------------------------------------------------------------------
__global__ __launch_bounds__(256)
void add_ln_kernel(const float* __restrict__ a, const float* __restrict__ r,
                   const float* __restrict__ gamma, const float* __restrict__ beta,
                   float* __restrict__ y, bf16* __restrict__ yb) {
  const long long row = blockIdx.x;
  const float* pa = a + row * Dn;
  const float* pr = r + row * Dn;
  const int tid = threadIdx.x;

  float s[4], sum = 0.0f, sq = 0.0f;
#pragma unroll
  for (int i = 0; i < 4; ++i) {
    int d = tid + i * 256;
    float v = pa[d] + pr[d];
    s[i] = v; sum += v; sq += v * v;
  }
  __shared__ float r1[256], r2[256];
  r1[tid] = sum; r2[tid] = sq; __syncthreads();
  for (int off = 128; off > 0; off >>= 1) {
    if (tid < off) { r1[tid] += r1[tid + off]; r2[tid] += r2[tid + off]; }
    __syncthreads();
  }
  float mean = r1[0] * (1.0f / Dn);
  float var  = r2[0] * (1.0f / Dn) - mean * mean;
  float inv  = rsqrtf(var + 1e-5f);
#pragma unroll
  for (int i = 0; i < 4; ++i) {
    int d = tid + i * 256;
    float v = (s[i] - mean) * inv * gamma[d] + beta[d];
    y[row * Dn + d] = v;
    if (yb) yb[row * Dn + d] = (bf16)v;
  }
}

// ---------------------------------------------------------------------------
// Host launcher
// ---------------------------------------------------------------------------
extern "C" void kernel_launch(void* const* d_in, const int* in_sizes, int n_in,
                              void* d_out, int out_size, void* d_ws, size_t ws_size,
                              hipStream_t stream) {
  (void)in_sizes; (void)n_in; (void)out_size; (void)ws_size;
  const float* x     = (const float*)d_in[0];
  const unsigned char* mask = (const unsigned char*)d_in[1];
  const float* Wq    = (const float*)d_in[2];
  const float* Wk    = (const float*)d_in[3];
  const float* Wv    = (const float*)d_in[4];
  const float* Wo    = (const float*)d_in[5];
  const float* W1    = (const float*)d_in[6];
  const float* b1    = (const float*)d_in[7];
  const float* W2    = (const float*)d_in[8];
  const float* b2    = (const float*)d_in[9];
  const float* gamma = (const float*)d_in[10];
  const float* beta  = (const float*)d_in[11];
  float* out = (float*)d_out;

  char* ws = (char*)d_ws;
  size_t off = 0;
  auto alloc = [&](size_t bytes) -> char* {
    char* p = ws + off;
    off = (off + bytes + 255) & ~(size_t)255;
    return p;
  };
  const long long BL = (long long)Bn * Ln;   // 8192

  bf16* xb   = (bf16*)alloc(BL * Dn * 2);
  bf16* WqT  = (bf16*)alloc((size_t)Dn * HKn * 2);   // HK x D
  bf16* WkT  = (bf16*)alloc((size_t)Dn * HKn * 2);   // HK x D
  bf16* WvT  = (bf16*)alloc((size_t)Dn * HVn * 2);   // HV x D
  bf16* WoT  = (bf16*)alloc((size_t)HVn * Dn * 2);   // D  x HV
  bf16* W1T  = (bf16*)alloc((size_t)Dn * Hid * 2);   // Hid x D
  bf16* W2T  = (bf16*)alloc((size_t)Hid * Dn * 2);   // D  x Hid
  bf16* q    = (bf16*)alloc(BL * HKn * 2);
  bf16* k    = (bf16*)alloc(BL * HKn * 2);
  bf16* vT   = (bf16*)alloc(BL * HVn * 2);           // (HV, B*L)
  float* sc  = (float*)alloc(BL * Ln * 4);
  bf16* attn = (bf16*)alloc(BL * Ln * 2);
  bf16* av   = (bf16*)alloc(BL * HVn * 2);
  float* zraw= (float*)alloc(BL * Dn * 4);
  float* z   = (float*)alloc(BL * Dn * 4);
  bf16* zb   = (bf16*)alloc(BL * Dn * 2);
  bf16* h    = (bf16*)alloc(BL * Hid * 2);
  float* fraw= (float*)alloc(BL * Dn * 4);

  // 1) x -> bf16 ; weights -> transposed bf16 (N-major)
  {
    long long n4 = BL * Dn / 4;
    cvt_f32_bf16<<<4096, 256, 0, stream>>>(x, xb, n4);
  }
  {
    dim3 b(32, 8);
    transpose_cvt_f32_bf16<<<dim3(HKn / 32, Dn / 32), b, 0, stream>>>(Wq, WqT, Dn, HKn);
    transpose_cvt_f32_bf16<<<dim3(HKn / 32, Dn / 32), b, 0, stream>>>(Wk, WkT, Dn, HKn);
    transpose_cvt_f32_bf16<<<dim3(HVn / 32, Dn / 32), b, 0, stream>>>(Wv, WvT, Dn, HVn);
    transpose_cvt_f32_bf16<<<dim3(Dn / 32, HVn / 32), b, 0, stream>>>(Wo, WoT, HVn, Dn);
    transpose_cvt_f32_bf16<<<dim3(Hid / 32, Dn / 32), b, 0, stream>>>(W1, W1T, Dn, Hid);
    transpose_cvt_f32_bf16<<<dim3(Dn / 32, Hid / 32), b, 0, stream>>>(W2, W2T, Hid, Dn);
  }

  // 2) Q, K projections (bf16 out), V projection stored transposed
  {
    dim3 g(HKn / 128, (int)(BL / 128), 1);
    gemm_bt_wmma<true, false, false, false, false><<<g, 256, 0, stream>>>(
        xb, WqT, q, nullptr, 1.0f, (int)BL, HKn, Dn, Dn, Dn, HKn, 0, 0, 0);
    gemm_bt_wmma<true, false, false, false, false><<<g, 256, 0, stream>>>(
        xb, WkT, k, nullptr, 1.0f, (int)BL, HKn, Dn, Dn, Dn, HKn, 0, 0, 0);
    gemm_bt_wmma<true, true, false, false, false><<<g, 256, 0, stream>>>(
        xb, WvT, vT, nullptr, 1.0f, (int)BL, HVn, Dn, Dn, Dn, (int)BL, 0, 0, 0);
  }

  // 3) scores = q @ k^T / sqrt(NUM_K)   (k is naturally N-major here)
  {
    const float alpha = 0.08838834764831845f; // 1/sqrt(128)
    dim3 g(Ln / 128, Ln / 128, Bn);
    gemm_bt_wmma<false, false, false, false, true><<<g, 256, 0, stream>>>(
        q, k, sc, nullptr, alpha, Ln, Ln, HKn, HKn, HKn, Ln,
        (long long)Ln * HKn, (long long)Ln * HKn, (long long)Ln * Ln);
  }

  // 4) masked softmax -> attn (bf16)
  {
    dim3 g(Ln, Bn);
    softmax_mask_kernel<<<g, 256, 0, stream>>>(sc, mask, attn);
  }

  // 5) attn @ v   (B = vT, per-batch column offset sB = L, ldb = B*L)
  {
    dim3 g(HVn / 128, Ln / 128, Bn);
    gemm_bt_wmma<true, false, false, false, false><<<g, 256, 0, stream>>>(
        attn, vT, av, nullptr, 1.0f, Ln, HVn, Ln, Ln, (int)BL, HVn,
        (long long)Ln * Ln, (long long)Ln, (long long)Ln * HVn);
  }

  // 6) zraw = (attn@v) @ Wo
  {
    dim3 g(Dn / 128, (int)(BL / 128), 1);
    gemm_bt_wmma<false, false, false, false, false><<<g, 256, 0, stream>>>(
        av, WoT, zraw, nullptr, 1.0f, (int)BL, Dn, HVn, HVn, HVn, Dn, 0, 0, 0);
  }

  // 7) z = LN(x + zraw)
  add_ln_kernel<<<(int)BL, 256, 0, stream>>>(x, zraw, gamma, beta, z, zb);

  // 8) h = relu(zb @ W1 + b1)
  {
    dim3 g(Hid / 128, (int)(BL / 128), 1);
    gemm_bt_wmma<true, false, true, true, false><<<g, 256, 0, stream>>>(
        zb, W1T, h, b1, 1.0f, (int)BL, Hid, Dn, Dn, Dn, Hid, 0, 0, 0);
  }

  // 9) fraw = h @ W2 + b2
  {
    dim3 g(Dn / 128, (int)(BL / 128), 1);
    gemm_bt_wmma<false, false, true, false, false><<<g, 256, 0, stream>>>(
        h, W2T, fraw, b2, 1.0f, (int)BL, Dn, Hid, Hid, Hid, Dn, 0, 0, 0);
  }

  // 10) out = LN(z + fraw)
  add_ln_kernel<<<(int)BL, 256, 0, stream>>>(z, fraw, gamma, beta, out, nullptr);
}
